// BRIGHT_54004918779983
// MI455X (gfx1250) — matrix operand order repaired and checked by reading, not compiled, not run on hardware
//
#include <hip/hip_runtime.h>
#include <stdint.h>

// ---------------------------------------------------------------------------
// Instant-NGP hash encoding + 2-layer MLP, fully fused for MI455X (gfx1250).
//   z:[N,3] f32, tables:[16,2^19,8] f32, W1:[128,128], b1:[128],
//   W2:[256,128], b2:[256]  ->  out:[N,256] f32
//
// Roofline: output stream = 512 MB (dominant); fusing the encoding into the
// MLP kernel keeps the [128 x 128] f16 encoding tile in LDS (32 KB) and
// removes the 256 MB HBM round-trip a separate encode pass would cost.
// GEMM FLOPs (51.5 G) run on v_wmma_f32_16x16x32_f16 -> compute is free.
// ---------------------------------------------------------------------------

typedef __attribute__((ext_vector_type(16))) _Float16 v16h;
typedef __attribute__((ext_vector_type(8)))  _Float16 v8h;
typedef __attribute__((ext_vector_type(8)))  float    v8f;
typedef __attribute__((ext_vector_type(4)))  float    v4f;

#define NPTS    524288
#define NLVL    16
#define TSIZE   524288      // 2^19 entries per level
#define FEAT    8
#define ENC_DIM 128
#define HID_DIM 128
#define OUT_DIM 256
#define LDH     136         // padded LDS row stride (halfs): 272B -> conflict-free

// floor(16 * 1.5^l) for l = 0..15 (matches reference exactly)
__constant__ float c_res[16] = {
    16.f, 24.f, 36.f, 54.f, 81.f, 121.f, 182.f, 273.f,
    410.f, 615.f, 922.f, 1383.f, 2075.f, 3113.f, 4670.f, 7006.f};

// ---------------------------------------------------------------------------
// Kernel 1: convert f32 weights to f16 so WMMA B-fragments are b128 loads.
// Weights stay L2-resident (96 KB) across the whole main kernel.
// ---------------------------------------------------------------------------
__global__ void BRIGHT_54004918779983_cvt(const float* __restrict__ W1,
                                          const float* __restrict__ W2,
                                          _Float16* __restrict__ W1h,
                                          _Float16* __restrict__ W2h) {
  int i = blockIdx.x * blockDim.x + threadIdx.x;
  if (i < HID_DIM * ENC_DIM) W1h[i] = (_Float16)W1[i];
  if (i < OUT_DIM * ENC_DIM) W2h[i] = (_Float16)W2[i];
}

// ---------------------------------------------------------------------------
// WMMA fragment helpers (16-bit A/B in contiguous k order k = 16h + s, wave32):
//  A 16x32:  lane<16 -> row M=lane, halfs K[0..15]
//            lane>=16 -> row M=lane-16, halfs K[16..31]
//            => one contiguous 32B run (two b128 loads).
//  B 32x16:  lane<16 -> col N=lane, K[0..15]; lane>=16 -> col, K[16..31]
//            B[k][n] = W[n][k] (row-major W) => 16 contiguous halfs of W's row.
//  C/D 16x16 f32: VGPR r, lane<16 -> (M=r, N=lane); lane>=16 -> (M=8+r, N=lane-16)
// ---------------------------------------------------------------------------
__device__ inline v16h pack16(v8h lo, v8h hi) {
  v16h r;
#pragma unroll
  for (int i = 0; i < 8; ++i) { r[i] = lo[i]; r[i + 8] = hi[i]; }
  return r;
}

__device__ inline v16h load_frag_a_lds(const _Float16* row, int c0) {
  v8h lo = *reinterpret_cast<const v8h*>(row + c0);        // ds_load_b128
  v8h hi = *reinterpret_cast<const v8h*>(row + c0 + 8);
  return pack16(lo, hi);
}

__device__ inline v16h load_frag_b(const _Float16* __restrict__ wrow, int k0) {
  v8h lo = *reinterpret_cast<const v8h*>(wrow + k0);       // global_load_b128
  v8h hi = *reinterpret_cast<const v8h*>(wrow + k0 + 8);
  return pack16(lo, hi);
}

// ---------------------------------------------------------------------------
// Kernel 2: fully fused encode + MLP. 256 threads = 8 wave32; 128 points/block.
//   Stage 0: 128 points x 16 levels hash-encoded cooperatively into LDS (f16).
//   Stage 1: h = LeakyReLU(enc @ W1^T + b1) via WMMA, staged to LDS (f16).
//   Stage 2: out = h @ W2^T + b2 via WMMA, streamed to global f32.
// Uniform control flow throughout: EXEC all-ones as WMMA requires.
// ---------------------------------------------------------------------------
__global__ __launch_bounds__(256)
void BRIGHT_54004918779983_fused(const float* __restrict__ z,
                                 const float* __restrict__ tables,
                                 const _Float16* __restrict__ W1h,
                                 const float* __restrict__ b1,
                                 const _Float16* __restrict__ W2h,
                                 const float* __restrict__ b2,
                                 float* __restrict__ out) {
  __shared__ _Float16 encT[128 * LDH];   // 34 KB encoding tile
  __shared__ _Float16 hT[128 * LDH];     // 34 KB hidden tile

  const int tid  = (int)threadIdx.x;
  const int wave = tid >> 5;             // 0..7
  const int lane = tid & 31;
  const int colL = lane & 15;            // N column (or M row for A frags)
  const int rsel = lane >> 4;            // lane-half select
  const int m0g  = (int)blockIdx.x * 128;
  const int mB   = wave * 16;            // wave's m-tile within block

  // ======== Stage 0: hash encode 128 points x 16 levels into LDS ========
  // Consecutive threads -> consecutive points at the same level: wave-local
  // gathers stay inside one 16 MB table level (L2-friendly).
#pragma unroll 1
  for (int s = 0; s < 8; ++s) {
    const int t     = tid + s * 256;     // 0..2047
    const int pl    = t & 127;           // point within block
    const int level = t >> 7;            // 0..15
    const int point = m0g + pl;

    const float zx = z[point * 3 + 0];
    const float zy = z[point * 3 + 1];
    const float zz = z[point * 3 + 2];
    const float res = c_res[level];
    const float fx = zx * res, fy = zy * res, fz = zz * res;
    const float x0f = floorf(fx), y0f = floorf(fy), z0f = floorf(fz);
    const float wx = fx - x0f, wy = fy - y0f, wz = fz - z0f;
    const uint32_t x0 = (uint32_t)x0f, y0 = (uint32_t)y0f, z0 = (uint32_t)z0f;

    const float* tab = tables + (size_t)level * (size_t)TSIZE * FEAT;

    float acc[FEAT];
#pragma unroll
    for (int i = 0; i < FEAT; ++i) acc[i] = 0.f;

#pragma unroll
    for (int c = 0; c < 8; ++c) {
      const uint32_t bx = (c >> 2) & 1u, by = (c >> 1) & 1u, bz = c & 1u;
      // wrap-around uint32 hash; PRIMES = [1, 2654435761, 805459861]
      const uint32_t h =
          (x0 + bx) * 1u ^ (y0 + by) * 2654435761u ^ (z0 + bz) * 805459861u;
      const uint32_t id = h & (uint32_t)(TSIZE - 1);
      const v4f* fp = reinterpret_cast<const v4f*>(tab + (size_t)id * FEAT);
      const v4f f0 = fp[0];
      const v4f f1 = fp[1];
      const float cw =
          (bx ? wx : 1.f - wx) * (by ? wy : 1.f - wy) * (bz ? wz : 1.f - wz);
#pragma unroll
      for (int i = 0; i < 4; ++i) {
        acc[i]     += cw * f0[i];
        acc[i + 4] += cw * f1[i];
      }
    }

    v8h o;
#pragma unroll
    for (int i = 0; i < FEAT; ++i) o[i] = (_Float16)acc[i];
    *reinterpret_cast<v8h*>(encT + pl * LDH + level * FEAT) = o;  // ds_store_b128
  }

  __syncthreads();

  // ======== Stage 1: layer 1 via WMMA ========
  // Wave w owns rows [16w, 16w+16): A-fragments loaded once from LDS and
  // reused across all 8 n-tiles.
  const _Float16* aRow1 = encT + (mB + colL) * LDH;
  v16h a1[4];
#pragma unroll
  for (int kk = 0; kk < 4; ++kk) a1[kk] = load_frag_a_lds(aRow1, kk * 32 + rsel * 16);

#pragma unroll
  for (int nt = 0; nt < 8; ++nt) {
    const int n0 = nt * 16;
    const _Float16* bRow = W1h + (size_t)(n0 + colL) * ENC_DIM + rsel * 16;
    v8f c = {};
#pragma unroll
    for (int kk = 0; kk < 4; ++kk) {
      v16h b = load_frag_b(bRow, kk * 32);
      c = __builtin_amdgcn_wmma_f32_16x16x32_f16(false, a1[kk], false, b,
                                                 (short)0, c, false, false);
    }
    const float bias = b1[n0 + colL];
#pragma unroll
    for (int r = 0; r < 8; ++r) {
      float v = c[r] + bias;
      v = (v >= 0.f) ? v : 0.01f * v;                       // LeakyReLU(0.01)
      hT[(mB + rsel * 8 + r) * LDH + n0 + colL] = (_Float16)v;
    }
  }

  __syncthreads();

  // ======== Stage 2: layer 2 via WMMA, stream f32 results to global ========
  const _Float16* aRow2 = hT + (mB + colL) * LDH;
  v16h a2[4];
#pragma unroll
  for (int kk = 0; kk < 4; ++kk) a2[kk] = load_frag_a_lds(aRow2, kk * 32 + rsel * 16);

#pragma unroll
  for (int nt = 0; nt < 16; ++nt) {
    const int n0 = nt * 16;
    const _Float16* bRow = W2h + (size_t)(n0 + colL) * HID_DIM + rsel * 16;
    v8f c = {};
#pragma unroll
    for (int kk = 0; kk < 4; ++kk) {
      v16h b = load_frag_b(bRow, kk * 32);
      c = __builtin_amdgcn_wmma_f32_16x16x32_f16(false, a2[kk], false, b,
                                                 (short)0, c, false, false);
    }
    const float bias = b2[n0 + colL];
    float* oBase = out + (size_t)(m0g + mB + rsel * 8) * OUT_DIM + n0 + colL;
#pragma unroll
    for (int r = 0; r < 8; ++r) oBase[(size_t)r * OUT_DIM] = c[r] + bias;
  }
}

// ---------------------------------------------------------------------------
extern "C" void kernel_launch(void* const* d_in, const int* in_sizes, int n_in,
                              void* d_out, int out_size, void* d_ws, size_t ws_size,
                              hipStream_t stream) {
  const float* z      = (const float*)d_in[0];
  const float* tables = (const float*)d_in[1];
  const float* W1     = (const float*)d_in[2];
  const float* b1     = (const float*)d_in[3];
  const float* W2     = (const float*)d_in[4];
  const float* b2     = (const float*)d_in[5];
  float* out = (float*)d_out;

  // Workspace: [W1h f16: 128*128] [W2h f16: 256*128]
  _Float16* W1h = (_Float16*)d_ws;
  _Float16* W2h = W1h + (size_t)HID_DIM * ENC_DIM;

  BRIGHT_54004918779983_cvt<<<(OUT_DIM * ENC_DIM + 255) / 256, 256, 0, stream>>>(
      W1, W2, W1h, W2h);

  BRIGHT_54004918779983_fused<<<NPTS / 128, 256, 0, stream>>>(
      z, tables, W1h, b1, W2h, b2, out);
}
